// GroundingNet_28613072126860
// MI455X (gfx1250) — matrix operand order repaired and checked by reading, not compiled, not run on hardware
//
#include <hip/hip_runtime.h>
#include <hip/hip_bf16.h>

#define BB   16
#define NN   64
#define EE   512
#define CIMG 512
#define H1C  256
#define H2C  128
#define H3C  64
#define UC   8
#define RC   8
#define EHC  64

typedef __attribute__((ext_vector_type(16))) _Float16 v16h;
typedef __attribute__((ext_vector_type(8)))  float    v8f;

__device__ __forceinline__ float sigm(float x) { return 1.f / (1.f + __expf(-x)); }

// ---- WMMA tile loaders (wave32), contiguous k = 16*(lane>>4) + h in both A and B ----
// A: 16x32 f16, lane = M row (lane&15); half h -> K = ((lane>=16)?16:0) + h
__device__ __forceinline__ v16h load_a_f32(const float* src, int ld, int lane) {
  v16h a;
  int m    = lane & 15;
  int koff = (lane >> 4) * 16;
#pragma unroll
  for (int h = 0; h < 16; ++h) {
    int k = koff + h;
    a[h] = (_Float16)src[m * ld + k];
  }
  return a;
}
// B: 32x16 f16, lane = N col (lane&15); half h -> K = ((lane>=16)?16:0) + h
__device__ __forceinline__ v16h load_b_f32(const float* w, int ld, int lane, int nmax) {
  v16h b;
  int n     = lane & 15;
  int kbase = (lane >> 4) * 16;
#pragma unroll
  for (int h = 0; h < 16; ++h) {
    int k = kbase + h;
    b[h] = (n < nmax) ? (_Float16)w[k * ld + n] : (_Float16)0.f;
  }
  return b;
}
#define WMMA_F16(A, Bm, C) \
  __builtin_amdgcn_wmma_f32_16x16x32_f16(false, (A), false, (Bm), (short)0, (C), false, false)

// ---------------- Kernel 1: build see + mask ----------------
__global__ __launch_bounds__(64) void build_see_kernel(const int* __restrict__ ei,
                                                       int* __restrict__ see,
                                                       float* __restrict__ mask) {
  __shared__ int deg[NN];
  int b = blockIdx.x;
  int t = threadIdx.x;
  deg[t] = 0;
  int* row = see + b * NN * NN + t * NN;
  row[0] = t;
#pragma unroll 4
  for (int k = 1; k < NN; ++k) row[k] = 0;
  __syncthreads();
  if (t == 0) {
    const int* src = ei + b * 2 * EE;
    const int* dst = src + EE;
    for (int e = 0; e < EE; ++e) {
      int d = dst[e];
      int s = src[e];
      int p = deg[d] + 1;   // pos = (#previous edges with same dst) + 1
      deg[d] = p;
      if (p < NN) see[b * NN * NN + d * NN + p] = s;
    }
  }
  __syncthreads();
  int cnt = deg[t] + 1;
  if (cnt > NN) cnt = NN;
  float* mrow = mask + b * NN * NN + t * NN;
#pragma unroll 4
  for (int k = 0; k < NN; ++k) mrow[k] = (k < cnt) ? 1.f : 0.f;
}

// ---------------- Kernel 2: fused node MLP (1 wave per 16-row tile) ----------------
__global__ __launch_bounds__(32) void node_mlp_kernel(
    const float* __restrict__ roi,
    const float* __restrict__ w1, const float* __restrict__ b1,
    const float* __restrict__ g1, const float* __restrict__ bt1,
    const float* __restrict__ m1, const float* __restrict__ v1,
    const float* __restrict__ w2, const float* __restrict__ b2,
    const float* __restrict__ g2, const float* __restrict__ bt2,
    const float* __restrict__ m2, const float* __restrict__ v2,
    const float* __restrict__ w3, const float* __restrict__ b3,
    const float* __restrict__ wn, const float* __restrict__ bnc,
    float* __restrict__ node_c) {
  __shared__ float h1[16 * H1C];
  __shared__ float h2[16 * H2C];
  __shared__ float h3[16 * H3C];
  int lane = threadIdx.x;
  int mt   = blockIdx.x;               // 0..63, rows mt*16 .. mt*16+15
  const float* x0 = roi + mt * 16 * CIMG;
  int nloc = lane & 15;
  int moff = (lane >> 4) * 8;

  // layer 1: 512 -> 256, BN + relu
  for (int nt = 0; nt < H1C / 16; ++nt) {
    v8f acc = {};
    for (int kt = 0; kt < CIMG / 32; ++kt) {
      v16h a  = load_a_f32(x0 + kt * 32, CIMG, lane);
      v16h bm = load_b_f32(w1 + kt * 32 * H1C + nt * 16, H1C, lane, 16);
      acc = WMMA_F16(a, bm, acc);
    }
    int n = nt * 16 + nloc;
    float rs = rsqrtf(v1[n] + 1e-5f);
    float bb = b1[n], gg = g1[n], be = bt1[n], mm = m1[n];
#pragma unroll
    for (int v = 0; v < 8; ++v) {
      float val = (acc[v] + bb - mm) * rs * gg + be;
      h1[(moff + v) * H1C + n] = fmaxf(val, 0.f);
    }
  }
  __syncthreads();
  // layer 2: 256 -> 128, BN + relu
  for (int nt = 0; nt < H2C / 16; ++nt) {
    v8f acc = {};
    for (int kt = 0; kt < H1C / 32; ++kt) {
      v16h a  = load_a_f32(h1 + kt * 32, H1C, lane);
      v16h bm = load_b_f32(w2 + kt * 32 * H2C + nt * 16, H2C, lane, 16);
      acc = WMMA_F16(a, bm, acc);
    }
    int n = nt * 16 + nloc;
    float rs = rsqrtf(v2[n] + 1e-5f);
    float bb = b2[n], gg = g2[n], be = bt2[n], mm = m2[n];
#pragma unroll
    for (int v = 0; v < 8; ++v) {
      float val = (acc[v] + bb - mm) * rs * gg + be;
      h2[(moff + v) * H2C + n] = fmaxf(val, 0.f);
    }
  }
  __syncthreads();
  // layer 3: 128 -> 64, bias + relu
  for (int nt = 0; nt < H3C / 16; ++nt) {
    v8f acc = {};
    for (int kt = 0; kt < H2C / 32; ++kt) {
      v16h a  = load_a_f32(h2 + kt * 32, H2C, lane);
      v16h bm = load_b_f32(w3 + kt * 32 * H3C + nt * 16, H3C, lane, 16);
      acc = WMMA_F16(a, bm, acc);
    }
    int n = nt * 16 + nloc;
    float bb = b3[n];
#pragma unroll
    for (int v = 0; v < 8; ++v) h3[(moff + v) * H3C + n] = fmaxf(acc[v] + bb, 0.f);
  }
  __syncthreads();
  // layer 4: 64 -> 8 (N padded to 16), sigmoid
  {
    v8f acc = {};
    for (int kt = 0; kt < H3C / 32; ++kt) {
      v16h a  = load_a_f32(h3 + kt * 32, H3C, lane);
      v16h bm = load_b_f32(wn + kt * 32 * UC, UC, lane, UC);
      acc = WMMA_F16(a, bm, acc);
    }
    if (nloc < UC) {
      float bb = bnc[nloc];
#pragma unroll
      for (int v = 0; v < 8; ++v)
        node_c[(mt * 16 + moff + v) * UC + nloc] = sigm(acc[v] + bb);
    }
  }
}

// ---------------- Kernel 3: P/Q rank-split of first edge layer ----------------
__global__ __launch_bounds__(256) void pq_kernel(const float* __restrict__ bbox,
                                                 const float* __restrict__ dir,
                                                 const float* __restrict__ pri,
                                                 const float* __restrict__ we1,
                                                 float* __restrict__ P,
                                                 float* __restrict__ Q) {
  int t = blockIdx.x * blockDim.x + threadIdx.x;
  if (t >= BB * NN * EHC) return;
  int row = t >> 6;
  int c   = t & 63;
  float a[9];
#pragma unroll
  for (int k = 0; k < 4; ++k) a[k] = bbox[row * 4 + k] * (1.f / 1024.f);
#pragma unroll
  for (int k = 0; k < 4; ++k) a[4 + k] = dir[row * 4 + k];
  a[8] = pri[row];
  float p = 0.f, q = 0.f;
#pragma unroll
  for (int k = 0; k < 9; ++k) {
    p += a[k] * we1[k * EHC + c];
    q += a[k] * we1[(9 + k) * EHC + c];
  }
  P[t] = p;
  Q[t] = q;
}

// ---------------- Kernel 4: edge MLP -> rel (1 wave per (b,i,j-tile)) ----------------
__global__ __launch_bounds__(32) void edge_mlp_kernel(
    const float* __restrict__ P, const float* __restrict__ Q,
    const float* __restrict__ be1v,
    const float* __restrict__ we2, const float* __restrict__ be2v,
    const float* __restrict__ wei, const float* __restrict__ beiv,
    float* __restrict__ rel) {
  __shared__ float e1[16 * EHC];
  __shared__ float e2[16 * EHC];
  int lane = threadIdx.x;
  int bt = blockIdx.x;                 // b*256 + i*4 + jt
  int jt = bt & 3;
  int i  = (bt >> 2) & 63;
  int b  = bt >> 8;
  const float* Pr = P + (b * NN + i) * EHC;
  // e1[jj][c] = relu(P[b,i,c] + Q[b,j,c] + be1[c]),  j = jt*16+jj
  for (int t = lane; t < 16 * EHC; t += 32) {
    int jj = t >> 6, c = t & 63;
    int j = jt * 16 + jj;
    e1[t] = fmaxf(Pr[c] + Q[(b * NN + j) * EHC + c] + be1v[c], 0.f);
  }
  __syncthreads();
  int nloc = lane & 15;
  int moff = (lane >> 4) * 8;
  // e2 = relu(e1 @ we2 + be2)   K=64, N=64
  for (int nt = 0; nt < 4; ++nt) {
    v8f acc = {};
    for (int kt = 0; kt < 2; ++kt) {
      v16h a  = load_a_f32(e1 + kt * 32, EHC, lane);
      v16h bm = load_b_f32(we2 + kt * 32 * EHC + nt * 16, EHC, lane, 16);
      acc = WMMA_F16(a, bm, acc);
    }
    int n = nt * 16 + nloc;
    float bb = be2v[n];
#pragma unroll
    for (int v = 0; v < 8; ++v) e2[(moff + v) * EHC + n] = fmaxf(acc[v] + bb, 0.f);
  }
  __syncthreads();
  // rel = sigmoid(e2 @ wei + bei)  K=64, N=8 (padded)
  {
    v8f acc = {};
    for (int kt = 0; kt < 2; ++kt) {
      v16h a  = load_a_f32(e2 + kt * 32, EHC, lane);
      v16h bm = load_b_f32(wei + kt * 32 * RC, RC, lane, RC);
      acc = WMMA_F16(a, bm, acc);
    }
    if (nloc < RC) {
      float bb = beiv[nloc];
#pragma unroll
      for (int v = 0; v < 8; ++v) {
        int j = jt * 16 + moff + v;
        rel[((b * NN + i) * NN + j) * RC + nloc] = sigm(acc[v] + bb);
      }
    }
  }
}

// ---------------- Kernel 5: node_concepts gather ----------------
__global__ __launch_bounds__(256) void node_gather_kernel(const int* __restrict__ see,
                                                          const float* __restrict__ mask,
                                                          const float* __restrict__ node_c,
                                                          float* __restrict__ out0) {
  int t = blockIdx.x * blockDim.x + threadIdx.x;   // (b*64+n)*64 + k
  if (t >= BB * NN * NN) return;
  int b = t >> 12;
  int s = see[t];
  float mk = mask[t];
  const float4* ncp = reinterpret_cast<const float4*>(node_c + (b * NN + s) * UC);
  float4 lo = ncp[0], hi = ncp[1];
  lo.x *= mk; lo.y *= mk; lo.z *= mk; lo.w *= mk;
  hi.x *= mk; hi.y *= mk; hi.z *= mk; hi.w *= mk;
  float4* op = reinterpret_cast<float4*>(out0 + (size_t)t * UC);
  op[0] = lo;
  op[1] = hi;
}

// ---------------- Kernel 6: edge_concepts gather (bandwidth-dominant, 134 MB) ----------------
__global__ __launch_bounds__(256) void edge_gather_kernel(const int* __restrict__ see,
                                                          const float* __restrict__ mask,
                                                          const float* __restrict__ rel,
                                                          float* __restrict__ out1) {
  int t = blockIdx.x * blockDim.x + threadIdx.x;   // m*4096 + j*64 + k, m=b*64+i
  if (t >= BB * NN * NN * NN) return;
  int m = t >> 12;
  int j = (t >> 6) & 63;
  int k = t & 63;
  int b = m >> 6;
  int sj = see[m * NN + j];
  int sk = see[m * NN + k];
  float mm = mask[m * NN + j] * mask[m * NN + k];
  const float4* rp =
      reinterpret_cast<const float4*>(rel + (size_t)((b * NN + sj) * NN + sk) * RC);
  float4 lo = rp[0], hi = rp[1];
  lo.x *= mm; lo.y *= mm; lo.z *= mm; lo.w *= mm;
  hi.x *= mm; hi.y *= mm; hi.z *= mm; hi.w *= mm;
  float4* op = reinterpret_cast<float4*>(out1 + (size_t)t * RC);
  op[0] = lo;
  op[1] = hi;
}

extern "C" void kernel_launch(void* const* d_in, const int* in_sizes, int n_in,
                              void* d_out, int out_size, void* d_ws, size_t ws_size,
                              hipStream_t stream) {
  const float* roi  = (const float*)d_in[0];
  const float* bbox = (const float*)d_in[1];
  const float* dir  = (const float*)d_in[2];
  const float* pri  = (const float*)d_in[3];
  const float* w1   = (const float*)d_in[4];
  const float* b1   = (const float*)d_in[5];
  const float* g1   = (const float*)d_in[6];
  const float* bt1  = (const float*)d_in[7];
  const float* m1   = (const float*)d_in[8];
  const float* v1   = (const float*)d_in[9];
  const float* w2   = (const float*)d_in[10];
  const float* b2   = (const float*)d_in[11];
  const float* g2   = (const float*)d_in[12];
  const float* bt2  = (const float*)d_in[13];
  const float* m2   = (const float*)d_in[14];
  const float* v2   = (const float*)d_in[15];
  const float* w3   = (const float*)d_in[16];
  const float* b3   = (const float*)d_in[17];
  const float* wn   = (const float*)d_in[18];
  const float* bnc  = (const float*)d_in[19];
  const float* we1  = (const float*)d_in[20];
  const float* be1  = (const float*)d_in[21];
  const float* we2  = (const float*)d_in[22];
  const float* be2  = (const float*)d_in[23];
  const float* wei  = (const float*)d_in[24];
  const float* bei  = (const float*)d_in[25];
  const int*   ei   = (const int*)d_in[26];

  float* ws    = (float*)d_ws;
  int*   see   = (int*)ws;            // 65536 ints
  float* mask  = ws + 65536;          // 65536 f
  float* ndc   = ws + 131072;         // 8192 f
  float* P     = ws + 139264;         // 65536 f
  float* Q     = ws + 204800;         // 65536 f
  float* rel   = ws + 270336;         // 524288 f   (total ~3.1 MB)

  float* out0 = (float*)d_out;                     // node_concepts: 524288 f
  float* out1 = out0 + (size_t)BB * NN * NN * UC;  // edge_concepts: 33554432 f

  build_see_kernel<<<BB, 64, 0, stream>>>(ei, see, mask);
  node_mlp_kernel<<<BB * NN / 16, 32, 0, stream>>>(roi, w1, b1, g1, bt1, m1, v1,
                                                   w2, b2, g2, bt2, m2, v2,
                                                   w3, b3, wn, bnc, ndc);
  pq_kernel<<<(BB * NN * EHC + 255) / 256, 256, 0, stream>>>(bbox, dir, pri, we1, P, Q);
  edge_mlp_kernel<<<BB * NN * (NN / 16), 32, 0, stream>>>(P, Q, be1, we2, be2, wei, bei, rel);
  node_gather_kernel<<<(BB * NN * NN + 255) / 256, 256, 0, stream>>>(see, mask, ndc, out0);
  edge_gather_kernel<<<(BB * NN * NN * NN + 255) / 256, 256, 0, stream>>>(see, mask, rel, out1);
}
